// MPGNN_65893388256021
// MI455X (gfx1250) — compile-verified, matrix operand order checked by reading, not run on hardware
//
#include <hip/hip_runtime.h>

// ---------------- problem constants (match reference) ----------------
#define N_NODES   40000
#define N_EDGES   640000
#define HID       128
#define N_LAYERS  4
#define NUM_GRAPHS 256

// ---------------- WMMA types & helpers ----------------
typedef __attribute__((ext_vector_type(16))) __bf16 v16bf;
typedef __attribute__((ext_vector_type(8)))  float  v8f;

union BF16Frag { uint4 u[2]; v16bf v; };

// float -> bf16, round-to-nearest-even (bit twiddle; no reliance on cvt builtins)
__device__ __forceinline__ __bf16 f2bf(float x) {
    union { float f; unsigned int u; } in; in.f = x;
    unsigned int r = in.u + 0x7fffu + ((in.u >> 16) & 1u);
    union { unsigned short s; __bf16 b; } out; out.s = (unsigned short)(r >> 16);
    return out.b;
}

// A fragment (16x32 bf16, MxK) from LDS, row-major bf16 with row stride strideE
// (strideE must be a multiple of 8 so the uint4 loads are 16B aligned).
// Per ISA layout: lanes 0-15: M=lane, K = 0..7 (elems 0..7), 16..23 (elems 8..15)
//                 lanes 16-31: M=lane-16, K = 8..15, 24..31
__device__ __forceinline__ v16bf load_a_frag_lds(const __bf16* As, int strideE,
                                                 int rowBase, int kb, int lane) {
    const int khalf = lane >> 4;
    const int row   = rowBase + (lane & 15);
    const __bf16* p = As + row * strideE + kb * 32 + khalf * 8;
    BF16Frag f;
    f.u[0] = *(const uint4*)(p);        // K  .. K+7
    f.u[1] = *(const uint4*)(p + 16);   // K+16 .. K+23
    return f.v;
}

// B fragment (32x16 bf16, KxN) pre-packed fragment-major: 512 bf16 per fragment,
// element (lane, j) at offset frag*512 + lane*16 + j  -> one contiguous 32B load/lane.
__device__ __forceinline__ v16bf load_b_frag(const __bf16* Bp, int frag, int lane) {
    const uint4* p = (const uint4*)(Bp + (size_t)frag * 512 + lane * 16);
    BF16Frag f;
    f.u[0] = p[0];
    f.u[1] = p[1];
    return f.v;
}

// ---------------- weight packing: f32 [K,128] -> bf16 fragment-major ----------------
// k mapping within a 32-K block for (lane, j):  k = 8*(lane>>4) + 16*(j>>3) + (j&7)
__global__ __launch_bounds__(256) void pack_weight_kernel(const float* __restrict__ W,
                                                          __bf16* __restrict__ out,
                                                          int Ksrc, int KB) {
    int idx = blockIdx.x * blockDim.x + threadIdx.x;
    int total = KB * 8 * 512;
    if (idx >= total) return;
    int f    = idx >> 9;       // fragment index = kb*8 + nb
    int r    = idx & 511;
    int lane = r >> 4;
    int j    = r & 15;
    int kb   = f >> 3;
    int nb   = f & 7;
    int n    = nb * 16 + (lane & 15);
    int k    = kb * 32 + ((lane >> 4) << 3) + ((j >> 3) << 4) + (j & 7);
    float v  = (k < Ksrc) ? W[(size_t)k * HID + n] : 0.0f;
    out[idx] = f2bf(v);
}

// ---------------- trivial kernels ----------------
__global__ __launch_bounds__(256) void zero_kernel(float* __restrict__ p, int n) {
    int i = blockIdx.x * blockDim.x + threadIdx.x;
    if (i < n) p[i] = 0.0f;
}

__global__ __launch_bounds__(256) void pool_kernel(const float* __restrict__ h,
                                                   const int* __restrict__ batch,
                                                   float* __restrict__ out, int n) {
    int i = blockIdx.x * blockDim.x + threadIdx.x;
    if (i < n) {
        int v = i >> 7;        // /128
        int c = i & 127;
        atomicAdd(&out[(size_t)batch[v] * HID + c], h[i]);
    }
}

// ---------------- generic single-output WMMA GEMM ----------------
// Out[M,128] = concat(A0[:, :w0], A1[:, :w1], zeros)[M,K] @ B[K,128] + bias
// K multiple of 32, M multiple of 64 preferred (guards included anyway).
// 256 threads = 8 waves: 4 in M (16 rows each) x 2 in N (64 cols each).
__global__ __launch_bounds__(256) void wmma_gemm_kernel(
    const float* __restrict__ A0, int w0, int ld0,
    const float* __restrict__ A1, int w1, int ld1,
    const __bf16* __restrict__ Bp, const float* __restrict__ bias,
    float* __restrict__ Out, int M, int K) {

    __shared__ __bf16 As[64 * 264];            // supports K up to 256 (stride K+8)
    const int tid   = threadIdx.x;
    const int mBase = blockIdx.x * 64;
    const int strideE = K + 8;                 // multiple of 8 -> 16B-aligned frags
    const int KB = K >> 5;

    // stage A tile (64 x K) as bf16 into LDS
    const int iters = (64 * K) >> 8;           // /256 threads
    for (int i = 0; i < iters; ++i) {
        int idx = tid + (i << 8);
        int row = idx / K;
        int k   = idx - row * K;
        int rg  = mBase + row;
        float v = 0.0f;
        if (rg < M) {
            if (k < w0)            v = A0[(size_t)rg * ld0 + k];
            else if (k < w0 + w1)  v = A1[(size_t)rg * ld1 + (k - w0)];
        }
        As[row * strideE + k] = f2bf(v);
    }
    __syncthreads();

    const int lane  = tid & 31;
    const int wid   = tid >> 5;
    const int waveM = wid & 3;
    const int waveN = wid >> 2;
    const int rowBase = waveM * 16;

    v8f c[4];
    #pragma unroll
    for (int t = 0; t < 4; ++t) {
        int col = waveN * 64 + t * 16 + (lane & 15);
        float b = bias[col];
        #pragma unroll
        for (int r = 0; r < 8; ++r) c[t][r] = b;
    }

    for (int kb = 0; kb < KB; ++kb) {
        v16bf a = load_a_frag_lds(As, strideE, rowBase, kb, lane);
        #pragma unroll
        for (int t = 0; t < 4; ++t) {
            int nb = waveN * 4 + t;
            v16bf b = load_b_frag(Bp, kb * 8 + nb, lane);
            c[t] = __builtin_amdgcn_wmma_f32_16x16x32_bf16(
                       false, a, false, b, (short)0, c[t], false, false);
        }
    }

    // C layout: elem r -> row = rowBase + 8*(lane>>4) + r, col = nb*16 + (lane&15)
    const int rowOff = rowBase + ((lane >> 4) << 3);
    #pragma unroll
    for (int t = 0; t < 4; ++t) {
        int col = waveN * 64 + t * 16 + (lane & 15);
        #pragma unroll
        for (int r = 0; r < 8; ++r) {
            int rg = mBase + rowOff + r;
            if (rg < M) Out[(size_t)rg * HID + col] = c[t][r];
        }
    }
}

// ---------------- fused edge kernel ----------------
// For a 64-edge tile:
//   cat3 = [h[send], h[rec], e]           (staged in LDS as bf16, 64x384)
//   msg  = cat3 @ Wm + bm  -> atomicAdd into agg[rec]
//   e    = cat3 @ We + be  -> written back IN PLACE (own rows only, read before write)
__global__ __launch_bounds__(256) void edge_mp_kernel(
    const float* __restrict__ h, float* __restrict__ e,
    const int* __restrict__ send, const int* __restrict__ rec,
    const __bf16* __restrict__ pWm, const float* __restrict__ bm,
    const __bf16* __restrict__ pWe, const float* __restrict__ be,
    float* __restrict__ agg, int E) {

    __shared__ __bf16 As[64 * 392];            // 64 x 384, stride 392 (mult of 8)
    __shared__ int sSend[64];
    __shared__ int sRec[64];

    const int tid   = threadIdx.x;
    const int eBase = blockIdx.x * 64;

    if (tid < 64) {
        int ed = eBase + tid;
        int s_ = 0, r_ = 0;
        if (ed < E) { s_ = send[ed]; r_ = rec[ed]; }
        sSend[tid] = s_;
        sRec[tid]  = r_;
    }
    __syncthreads();

    // stage 64 x 384 cat3 tile as bf16
    #pragma unroll 4
    for (int i = 0; i < 96; ++i) {
        int idx = tid + (i << 8);
        int row = idx / 384;
        int k   = idx - row * 384;
        int ed  = eBase + row;
        float v = 0.0f;
        if (ed < E) {
            if (k < 128)       v = h[(size_t)sSend[row] * HID + k];
            else if (k < 256)  v = h[(size_t)sRec[row]  * HID + (k - 128)];
            else               v = e[(size_t)ed * HID + (k - 256)];
        }
        As[row * 392 + k] = f2bf(v);
    }
    __syncthreads();

    const int lane  = tid & 31;
    const int wid   = tid >> 5;
    const int waveM = wid & 3;
    const int waveN = wid >> 2;
    const int rowBase = waveM * 16;

    v8f cm[4], ce[4];
    #pragma unroll
    for (int t = 0; t < 4; ++t) {
        int col = waveN * 64 + t * 16 + (lane & 15);
        float vm = bm[col], veb = be[col];
        #pragma unroll
        for (int r = 0; r < 8; ++r) { cm[t][r] = vm; ce[t][r] = veb; }
    }

    #pragma unroll
    for (int kb = 0; kb < 12; ++kb) {
        v16bf a = load_a_frag_lds(As, 392, rowBase, kb, lane);
        #pragma unroll
        for (int t = 0; t < 4; ++t) {
            int nb = waveN * 4 + t;
            v16bf bmf = load_b_frag(pWm, kb * 8 + nb, lane);
            cm[t] = __builtin_amdgcn_wmma_f32_16x16x32_bf16(
                        false, a, false, bmf, (short)0, cm[t], false, false);
            v16bf bef = load_b_frag(pWe, kb * 8 + nb, lane);
            ce[t] = __builtin_amdgcn_wmma_f32_16x16x32_bf16(
                        false, a, false, bef, (short)0, ce[t], false, false);
        }
    }

    const int rowOff = rowBase + ((lane >> 4) << 3);
    #pragma unroll
    for (int t = 0; t < 4; ++t) {
        int col = waveN * 64 + t * 16 + (lane & 15);
        #pragma unroll
        for (int r = 0; r < 8; ++r) {
            int rl = rowOff + r;
            int ed = eBase + rl;
            if (ed < E) {
                e[(size_t)ed * HID + col] = ce[t][r];                     // in-place edge update
                atomicAdd(&agg[(size_t)sRec[rl] * HID + col], cm[t][r]);  // scatter messages
            }
        }
    }
}

// ---------------- host launch ----------------
extern "C" void kernel_launch(void* const* d_in, const int* in_sizes, int n_in,
                              void* d_out, int out_size, void* d_ws, size_t ws_size,
                              hipStream_t stream) {
    (void)in_sizes; (void)n_in; (void)out_size; (void)ws_size;

    const float* h_in     = (const float*)d_in[0];   // [N,128]
    const float* e_in     = (const float*)d_in[1];   // [E,16]
    const int*   eidx     = (const int*)  d_in[2];   // [2,E]
    const int*   batch    = (const int*)  d_in[3];   // [N]
    const float* W_embed  = (const float*)d_in[4];
    const float* b_embed  = (const float*)d_in[5];
    const float* W_eembed = (const float*)d_in[6];
    const float* b_eembed = (const float*)d_in[7];
    const float* Wm       = (const float*)d_in[8];   // [4,384,128]
    const float* bm       = (const float*)d_in[9];   // [4,128]
    const float* Wh       = (const float*)d_in[10];  // [4,256,128]
    const float* bh       = (const float*)d_in[11];  // [4,128]
    const float* We       = (const float*)d_in[12];  // [4,384,128]
    const float* be       = (const float*)d_in[13];  // [4,128]
    float* out = (float*)d_out;                      // [256,128]

    const int* send = eidx;
    const int* rec  = eidx + N_EDGES;

    // workspace carve-up (256B aligned)
    char* ws = (char*)d_ws;
    size_t off = 0;
    auto carve = [&](size_t bytes) -> void* {
        void* p = ws + off;
        off = (off + bytes + 255) & ~(size_t)255;
        return p;
    };
    float*  hbuf     = (float*) carve((size_t)N_NODES * HID * 4);   // ~20.5 MB
    float*  ebuf     = (float*) carve((size_t)N_EDGES * HID * 4);   // ~328 MB
    float*  agg      = (float*) carve((size_t)N_NODES * HID * 4);   // ~20.5 MB
    __bf16* pWembed  = (__bf16*)carve((size_t)4  * 8 * 512 * 2);
    __bf16* pWeembed = (__bf16*)carve((size_t)1  * 8 * 512 * 2);
    __bf16* pWm      = (__bf16*)carve((size_t)N_LAYERS * 12 * 8 * 512 * 2);
    __bf16* pWe      = (__bf16*)carve((size_t)N_LAYERS * 12 * 8 * 512 * 2);
    __bf16* pWh      = (__bf16*)carve((size_t)N_LAYERS * 8  * 8 * 512 * 2);

    // 1) pack weights to bf16 fragment-major
    pack_weight_kernel<<<(4 * 8 * 512) / 256, 256, 0, stream>>>(W_embed,  pWembed,  128, 4);
    pack_weight_kernel<<<(1 * 8 * 512) / 256, 256, 0, stream>>>(W_eembed, pWeembed, 16,  1);
    for (int l = 0; l < N_LAYERS; ++l) {
        pack_weight_kernel<<<(12 * 8 * 512) / 256, 256, 0, stream>>>(
            Wm + (size_t)l * 384 * HID, pWm + (size_t)l * 12 * 8 * 512, 384, 12);
        pack_weight_kernel<<<(12 * 8 * 512) / 256, 256, 0, stream>>>(
            We + (size_t)l * 384 * HID, pWe + (size_t)l * 12 * 8 * 512, 384, 12);
        pack_weight_kernel<<<(8 * 8 * 512) / 256, 256, 0, stream>>>(
            Wh + (size_t)l * 256 * HID, pWh + (size_t)l * 8 * 8 * 512, 256, 8);
    }

    const int nodeBlocks = (N_NODES + 63) / 64;   // 625
    const int edgeBlocks = (N_EDGES + 63) / 64;   // 10000

    // 2) embeddings
    wmma_gemm_kernel<<<nodeBlocks, 256, 0, stream>>>(
        h_in, 128, 128, (const float*)nullptr, 0, 0,
        pWembed, b_embed, hbuf, N_NODES, 128);
    wmma_gemm_kernel<<<edgeBlocks, 256, 0, stream>>>(
        e_in, 16, 16, (const float*)nullptr, 0, 0,
        pWeembed, b_eembed, ebuf, N_EDGES, 32);   // K padded 16 -> 32

    // 3) message-passing layers
    const int aggN = N_NODES * HID;
    for (int l = 0; l < N_LAYERS; ++l) {
        zero_kernel<<<(aggN + 255) / 256, 256, 0, stream>>>(agg, aggN);
        edge_mp_kernel<<<edgeBlocks, 256, 0, stream>>>(
            hbuf, ebuf, send, rec,
            pWm + (size_t)l * 12 * 8 * 512, bm + (size_t)l * HID,
            pWe + (size_t)l * 12 * 8 * 512, be + (size_t)l * HID,
            agg, N_EDGES);
        wmma_gemm_kernel<<<nodeBlocks, 256, 0, stream>>>(
            hbuf, 128, 128, agg, 128, 128,
            pWh + (size_t)l * 8 * 8 * 512, bh + (size_t)l * HID,
            hbuf, N_NODES, 256);                  // h updated in place
    }

    // 4) global add pool
    const int outN = NUM_GRAPHS * HID;
    zero_kernel<<<(outN + 255) / 256, 256, 0, stream>>>(out, outN);
    pool_kernel<<<(aggN + 255) / 256, 256, 0, stream>>>(hbuf, batch, out, aggN);
}
